// RecLinear_32564442038393
// MI455X (gfx1250) — matrix operand order repaired and checked by reading, not compiled, not run on hardware
//
#include <hip/hip_runtime.h>
#include <math.h>

typedef unsigned int   u32;
typedef unsigned short u16;
typedef float v8f  __attribute__((ext_vector_type(8)));
typedef u32   v8ui __attribute__((ext_vector_type(8)));
typedef __bf16 v16bf __attribute__((ext_vector_type(16)));

#define XMINF  (-1.21f)
#define RANGEF (25.12f)   /* XMAX - XMIN */

static inline unsigned cdiv_h(size_t a, size_t b) { return (unsigned)((a + b - 1) / b); }

// ---------------------------------------------------------------- device utils
__device__ __forceinline__ u16 f2bf(float f) {
    u32 u = __float_as_uint(f);
    u += 0x7FFFu + ((u >> 16) & 1u);           // round-to-nearest-even
    return (u16)(u >> 16);
}
__device__ __forceinline__ float gelu_f(float x) {
    float x3 = x * x * x;
    return 0.5f * x * (1.0f + tanhf(0.7978845608028654f * (x + 0.044715f * x3)));
}
__device__ __forceinline__ u32 pcg(u32 x) {
    x = x * 747796405u + 2891336453u;
    u32 w = ((x >> ((x >> 28) + 4u)) ^ x) * 277803737u;
    return (w >> 22) ^ w;
}
__device__ __forceinline__ float u01(u32 h) { return (float)(h >> 8) * (1.0f / 16777216.0f); }
__device__ __forceinline__ float d_ratio() { return 0.8f + 0.4f * u01(pcg(0x2Au)); }
__device__ __forceinline__ u32 enc_ord(float f) {          // order-preserving float->uint
    u32 u = __float_as_uint(f);
    return (u & 0x80000000u) ? ~u : (u | 0x80000000u);
}
__device__ __forceinline__ float dec_ord(u32 u) {
    return (u & 0x80000000u) ? __uint_as_float(u & 0x7FFFFFFFu) : __uint_as_float(~u);
}

// ---------------------------------------------------------------- fill / convert
__global__ __launch_bounds__(256) void k_fill_u32(u32* p, u32 v, size_t n) {
    size_t i = (size_t)blockIdx.x * 256 + threadIdx.x;
    if (i < n) p[i] = v;
}
__global__ __launch_bounds__(256) void k_f2bf(const float* __restrict__ in, u16* __restrict__ out, size_t n) {
    size_t i = (size_t)blockIdx.x * 256 + threadIdx.x;
    if (i < n) out[i] = f2bf(in[i]);
}

// ---------------------------------------------------------------- WMMA GEMM
// out[M,Hout](+colOff) = act( A[M,K](bf16) @ W[K,Hout](f32->bf16 in LDS) + bias )
// One wave computes a 16-row x (HT*16)-col tile. Block = 8 waves = 128 rows.
// HT is a COMPILE-TIME template parameter so acc[] stays in statically-named
// VGPRs (runtime HT previously forced v_movrel dynamic VGPR indexing).
// B kept column-major in LDS: wlds[n*K + k], so each lane streams contiguous b128s.
template <typename OutT, int ACT, int HT>
__global__ __launch_bounds__(256) void wmma_gemm_kernel(
    const u16* __restrict__ A, int M, int K,
    const float* __restrict__ W, const float* __restrict__ bias,
    int Hout,
    OutT* __restrict__ out, int outStride, int outColOff)
{
    extern __shared__ u16 wlds[];              // [HT*16][K]
    const int tid  = threadIdx.x;
    const int Hpad = HT * 16;
    for (int i = tid; i < K * Hpad; i += 256) {
        int n = i % Hpad, k = i / Hpad;
        float w = (n < Hout) ? W[(size_t)k * Hout + n] : 0.0f;
        wlds[(size_t)n * K + k] = f2bf(w);
    }
    __syncthreads();

    const int wave = tid >> 5, lane = tid & 31;
    const int m0 = (blockIdx.x * 8 + wave) * 16;
    if (m0 >= M) return;

    int row = m0 + (lane & 15);
    if (row >= M) row = M - 1;                  // clamp; masked at store
    const u16* arow = A + (size_t)row * K;
    const int kb   = (lane >> 4) * 8;           // A: halves 0..7 -> K+kb, 8..15 -> K+16+kb
    const int ks   = (lane >> 4) * 8;           // B: halves 0..7 -> K+ks, 8..15 -> K+16+ks
    const int nloc = lane & 15;

    v8f acc[HT];
#pragma unroll
    for (int t = 0; t < HT; ++t) acc[t] = (v8f)(0.0f);

    for (int k0 = 0; k0 < K; k0 += 32) {
        const uint4 qa0 = *reinterpret_cast<const uint4*>(arow + k0 + kb);
        const uint4 qa1 = *reinterpret_cast<const uint4*>(arow + k0 + 16 + kb);
        v8ui ai;
        ai[0] = qa0.x; ai[1] = qa0.y; ai[2] = qa0.z; ai[3] = qa0.w;
        ai[4] = qa1.x; ai[5] = qa1.y; ai[6] = qa1.z; ai[7] = qa1.w;
        v16bf av = __builtin_bit_cast(v16bf, ai);
#pragma unroll
        for (int t = 0; t < HT; ++t) {
            const u16* bp = wlds + (size_t)(t * 16 + nloc) * K + k0 + ks;
            const uint4 qb0 = *reinterpret_cast<const uint4*>(bp);
            const uint4 qb1 = *reinterpret_cast<const uint4*>(bp + 16);
            v8ui bi;
            bi[0] = qb0.x; bi[1] = qb0.y; bi[2] = qb0.z; bi[3] = qb0.w;
            bi[4] = qb1.x; bi[5] = qb1.y; bi[6] = qb1.z; bi[7] = qb1.w;
            v16bf bv = __builtin_bit_cast(v16bf, bi);
            acc[t] = __builtin_amdgcn_wmma_f32_16x16x32_bf16(
                false, av, false, bv, (short)0, acc[t], false, false);
        }
    }

    const int rbase = (lane >> 4) * 8;          // C/D: lanes>=16 hold rows M=8..15
#pragma unroll
    for (int t = 0; t < HT; ++t) {
        int colg = t * 16 + nloc;
        if (colg >= Hout) continue;
        float bv = bias ? bias[colg] : 0.0f;
#pragma unroll
        for (int v = 0; v < 8; ++v) {
            int m = m0 + rbase + v;
            if (m >= M) continue;
            float val = acc[t][v] + bv;
            if (ACT == 1) val = gelu_f(val);
            size_t o = (size_t)m * outStride + outColOff + colg;
            if (sizeof(OutT) == 2) out[o] = (OutT)f2bf(val);
            else                   out[o] = (OutT)val;
        }
    }
}

// ---------------------------------------------------------------- small-K projection (H=64)
// out = act( X[M,K] @ W[K,64] + b )   act: 0 none, 1 gelu, 2 leaky(slope)
__global__ __launch_bounds__(256) void k_lin_small(
    const float* __restrict__ X, int M, int K,
    const float* __restrict__ W, const float* __restrict__ b,
    int act, float slope, u16* out_bf, float* out_f)
{
    size_t idx = (size_t)blockIdx.x * 256 + threadIdx.x;
    int e = (int)(idx >> 6), j = (int)(idx & 63);
    if (e >= M) return;
    float s = b[j];
    const float* xr = X + (size_t)e * K;
    for (int k = 0; k < K; ++k) s += xr[k] * W[k * 64 + j];
    if (act == 1) s = gelu_f(s);
    else if (act == 2) s = (s > 0.0f) ? s : slope * s;
    if (out_bf) out_bf[idx] = f2bf(s);
    else        out_f[idx] = s;
}

// ---------------------------------------------------------------- GATv2 edge kernels
__global__ __launch_bounds__(256) void k_message(
    const float* __restrict__ xl, const float* __restrict__ xr,
    const float* __restrict__ eattr, const int* __restrict__ src,
    const int* __restrict__ dst, const float* __restrict__ att,
    float* __restrict__ logit, u32* __restrict__ nmax, int E)
{
    int e = blockIdx.x * 256 + threadIdx.x;
    if (e >= E) return;
    int s = src[e], d = dst[e];
    const float* pl = xl + (size_t)s * 64;
    const float* pr = xr + (size_t)d * 64;
    const float* pe = eattr + (size_t)e * 64;
    float acc = 0.0f;
#pragma unroll 8
    for (int j = 0; j < 64; ++j) {
        float v = pl[j] + pr[j] + pe[j];
        v = (v > 0.0f) ? v : 0.2f * v;
        acc += v * att[j];
    }
    logit[e] = acc;
    atomicMax(&nmax[d], enc_ord(acc));
}

__global__ __launch_bounds__(256) void k_expsum(
    float* __restrict__ logit, const int* __restrict__ dst,
    const u32* __restrict__ nmax, float* __restrict__ nsum, int E)
{
    int e = blockIdx.x * 256 + threadIdx.x;
    if (e >= E) return;
    int d = dst[e];
    float m = dec_ord(nmax[d]);
    if (!isfinite(m)) m = 0.0f;
    float ev = expf(logit[e] - m);
    logit[e] = ev;                              // in-place: logit -> exp value
    atomicAdd(&nsum[d], ev);
}

__global__ __launch_bounds__(256) void k_scatter(
    const float* __restrict__ expv, const float* __restrict__ xl,
    const int* __restrict__ src, const int* __restrict__ dst,
    const float* __restrict__ nsum, float* __restrict__ acc, int E)
{
    size_t idx = (size_t)blockIdx.x * 256 + threadIdx.x;
    int e = (int)(idx >> 6), j = (int)(idx & 63);
    if (e >= E) return;
    int d = dst[e], s = src[e];
    float a = expv[e] / (nsum[d] + 1e-16f);
    atomicAdd(&acc[(size_t)d * 64 + j], xl[(size_t)s * 64 + j] * a);
}

__global__ __launch_bounds__(256) void k_combine(
    const float* __restrict__ acc, const float* __restrict__ bias,
    const float* __restrict__ pre, float* __restrict__ out, int N)
{
    size_t idx = (size_t)blockIdx.x * 256 + threadIdx.x;
    if (idx >= (size_t)N * 64) return;
    int j = (int)(idx & 63);
    out[idx] = gelu_f(acc[idx] + bias[j]) + pre[idx];
}

// ---------------------------------------------------------------- gathers
__global__ __launch_bounds__(256) void k_gather_cat2(
    const float* __restrict__ feat, const int* __restrict__ src,
    const int* __restrict__ dst, u16* __restrict__ cat2, int E)
{
    size_t idx = (size_t)blockIdx.x * 256 + threadIdx.x;
    int e = (int)(idx >> 7), c = (int)(idx & 127);
    if (e >= E) return;
    int node = (c < 64) ? src[e] : dst[e];
    cat2[idx] = f2bf(feat[(size_t)node * 64 + (c & 63)]);
}

__global__ __launch_bounds__(256) void k_gather_misc(
    const int* __restrict__ index, const int* __restrict__ cur_t,
    const int* __restrict__ cur_w, const float* __restrict__ embed,
    const float* __restrict__ ttab, const float* __restrict__ wtab,
    u16* __restrict__ xf6, int E)
{
    size_t idx = (size_t)blockIdx.x * 256 + threadIdx.x;
    int e = (int)(idx >> 6), j = (int)(idx & 63);
    if (e >= E) return;
    size_t base = (size_t)e * 384;
    xf6[base + j]       = f2bf(embed[(size_t)index[e] * 64 + j]);
    xf6[base + 256 + j] = f2bf(ttab[(size_t)cur_t[e] * 64 + j]);
    xf6[base + 320 + j] = f2bf(wtab[(size_t)cur_w[e] * 64 + j]);
}

// ---------------------------------------------------------------- VAE branch
__global__ __launch_bounds__(256) void k_vae_fc1(
    const float* __restrict__ x, const float* __restrict__ fc1_W,
    float* __restrict__ hraw, int N)
{
    __shared__ float xs[256];
    const int j = threadIdx.x;
    int chunk = (N + gridDim.x - 1) / gridDim.x;
    int n0 = blockIdx.x * chunk, n1 = min(N, n0 + chunk);
    const float ratio = d_ratio();
    const float invR = 1.0f / RANGEF;
    float acc[4] = {0.f, 0.f, 0.f, 0.f};
    for (int nb = n0; nb < n1; nb += 64) {
        int cnt = min(64, n1 - nb);
        __syncthreads();
        if (j < cnt * 4) xs[j] = x[(size_t)nb * 4 + j];
        __syncthreads();
        for (int r = 0; r < cnt; ++r) {
            int n = nb + r;
            float drop = (u01(pcg(1000u + (u32)n)) > 0.4f) ? 1.0f : 0.0f;
            float fac = ratio * drop * invR;
            float w = fc1_W[(size_t)n * 256 + j];
#pragma unroll
            for (int i = 0; i < 4; ++i) acc[i] += (xs[r * 4 + i] - XMINF) * fac * w;
        }
    }
#pragma unroll
    for (int i = 0; i < 4; ++i) atomicAdd(&hraw[i * 256 + j], acc[i]);
}

__global__ __launch_bounds__(256) void k_vae_head(
    const float* __restrict__ hraw, const float* __restrict__ fc1_b,
    const float* __restrict__ fc2_W, const float* __restrict__ fc2_b,
    const float* __restrict__ fc3_W, const float* __restrict__ fc3_b,
    const float* __restrict__ fc4_W, const float* __restrict__ fc4_b,
    float* __restrict__ g)
{
    __shared__ float hh[1024];
    __shared__ float zz[128];
    int j = threadIdx.x;
#pragma unroll
    for (int i = 0; i < 4; ++i) hh[i * 256 + j] = gelu_f(hraw[i * 256 + j] + fc1_b[j]);
    __syncthreads();
    if (j < 128) {
        int i = j >> 5, c = j & 31;
        float mu = fc2_b[c], lv = fc3_b[c];
        for (int k = 0; k < 256; ++k) {
            float h = hh[i * 256 + k];
            mu += h * fc2_W[k * 32 + c];
            lv += h * fc3_W[k * 32 + c];
        }
        float uu1 = u01(pcg(9000u + (u32)j)); if (uu1 < 1e-7f) uu1 = 1e-7f;
        float uu2 = u01(pcg(9500u + (u32)j));
        float nrm = sqrtf(-2.0f * logf(uu1)) * cosf(6.28318530718f * uu2);
        zz[i * 32 + c] = mu + nrm * expf(lv * 0.5f);
    }
    __syncthreads();
#pragma unroll
    for (int i = 0; i < 4; ++i) {
        float s = fc4_b[j];
        for (int k = 0; k < 32; ++k) s += zz[i * 32 + k] * fc4_W[k * 256 + j];
        g[i * 256 + j] = gelu_f(s);
    }
}

__global__ __launch_bounds__(256) void k_vae_rec(
    const float* __restrict__ g, const float* __restrict__ fc5_W,
    const float* __restrict__ fc5_b, const float* __restrict__ x,
    float* __restrict__ out_rec, float* __restrict__ xrec1, int N)
{
    __shared__ float gs[1024];
    for (int i = threadIdx.x; i < 1024; i += 256) gs[i] = g[i];
    __syncthreads();
    int n = blockIdx.x * 256 + threadIdx.x;
    if (n >= N) return;
    float a0 = 0, a1 = 0, a2 = 0, a3 = 0;
    for (int k = 0; k < 256; ++k) {
        float w = fc5_W[(size_t)k * N + n];
        a0 += gs[k] * w; a1 += gs[256 + k] * w; a2 += gs[512 + k] * w; a3 += gs[768 + k] * w;
    }
    float ratio = d_ratio(), b = fc5_b[n];
    float xr[4];
    xr[0] = (a0 + b) / ratio * RANGEF + XMINF;
    xr[1] = (a1 + b) / ratio * RANGEF + XMINF;
    xr[2] = (a2 + b) / ratio * RANGEF + XMINF;
    xr[3] = (a3 + b) / ratio * RANGEF + XMINF;
    float sx = x[(size_t)n * 4] + x[(size_t)n * 4 + 1] + x[(size_t)n * 4 + 2] + x[(size_t)n * 4 + 3];
    float mask = (sx != XMINF * 4.0f) ? 1.0f : 0.0f;
#pragma unroll
    for (int i = 0; i < 4; ++i) {
        out_rec[(size_t)n * 4 + i] = xr[i];
        xrec1[(size_t)n * 4 + i] = mask * x[(size_t)n * 4 + i] + (1.0f - mask) * xr[i];
    }
}

// ---------------------------------------------------------------- host
extern "C" void kernel_launch(void* const* d_in, const int* in_sizes, int n_in,
                              void* d_out, int out_size, void* d_ws, size_t ws_size,
                              hipStream_t stream)
{
    const int*   index_t  = (const int*)d_in[0];
    const int*   eidx     = (const int*)d_in[1];
    const float* x        = (const float*)d_in[2];
    const float* attr     = (const float*)d_in[3];
    const int*   cur_t    = (const int*)d_in[4];
    const int*   cur_w    = (const int*)d_in[5];
    const float* embed    = (const float*)d_in[6];
    const float* node_emb = (const float*)d_in[7];
    const float* ttab     = (const float*)d_in[8];
    const float* wtab     = (const float*)d_in[9];
    const float* node_lin_W = (const float*)d_in[10];
    const float* node_lin_b = (const float*)d_in[11];
    const float* attr_lin_W = (const float*)d_in[12];
    const float* attr_lin_b = (const float*)d_in[13];
    const float* attr1_W1 = (const float*)d_in[14];
    const float* attr1_b1 = (const float*)d_in[15];
    const float* attr1_W2 = (const float*)d_in[16];
    const float* attr1_b2 = (const float*)d_in[17];
    const float* lins_W0  = (const float*)d_in[18];
    const float* lins_b0  = (const float*)d_in[19];
    const float* lins_W1  = (const float*)d_in[20];
    const float* lins_b1  = (const float*)d_in[21];
    const float* lins_W2  = (const float*)d_in[22];
    const float* lins_b2  = (const float*)d_in[23];
    const float* fc1_W = (const float*)d_in[24];
    const float* fc1_b = (const float*)d_in[25];
    const float* fc2_W = (const float*)d_in[26];
    const float* fc2_b = (const float*)d_in[27];
    const float* fc3_W = (const float*)d_in[28];
    const float* fc3_b = (const float*)d_in[29];
    const float* fc4_W = (const float*)d_in[30];
    const float* fc4_b = (const float*)d_in[31];
    const float* fc5_W = (const float*)d_in[32];
    const float* fc5_b = (const float*)d_in[33];
    const float* gcn1_W = (const float*)d_in[34];
    const float* gcn1_b = (const float*)d_in[35];
    const float* gcn2_W = (const float*)d_in[36];
    const float* gcn2_b = (const float*)d_in[37];
    const float* cW[2][3] = {{(const float*)d_in[38], (const float*)d_in[40], (const float*)d_in[42]},
                             {(const float*)d_in[45], (const float*)d_in[47], (const float*)d_in[49]}};
    // cW[s] = {Wl, Wr, We}
    const float* cBl[2] = {(const float*)d_in[39], (const float*)d_in[46]};
    const float* cBr[2] = {(const float*)d_in[41], (const float*)d_in[48]};
    const float* cAtt[2]  = {(const float*)d_in[43], (const float*)d_in[50]};
    const float* cBias[2] = {(const float*)d_in[44], (const float*)d_in[51]};

    const int E = in_sizes[0];
    const int N = in_sizes[7] / 64;
    const int* src = eidx;
    const int* dst = eidx + E;

    float* out_xf  = (float*)d_out;
    float* out_rec = out_xf + (size_t)E * 8;

    // -------- workspace bump allocator
    unsigned char* wsp = (unsigned char*)d_ws;
    size_t off = 0;
    auto walloc = [&](size_t bytes) -> void* {
        void* p = wsp + off;
        off += (bytes + 255) & ~(size_t)255;
        return p;
    };
    u16*   attr1_bf = (u16*)walloc((size_t)E * 64 * 2);
    float* eattr    = (float*)walloc((size_t)E * 64 * 4);
    float* edge_s   = (float*)walloc((size_t)E * 4);
    u32*   nmax     = (u32*)walloc((size_t)N * 4);
    float* nsum     = (float*)walloc((size_t)N * 4);
    float* nacc     = (float*)walloc((size_t)N * 64 * 4);
    float* ne_cur   = (float*)walloc((size_t)N * 64 * 4);
    float* data_cur = (float*)walloc((size_t)N * 64 * 4);
    float* xl       = (float*)walloc((size_t)N * 64 * 4);
    float* xr       = (float*)walloc((size_t)N * 64 * 4);
    u16*   nbf      = (u16*)walloc((size_t)N * 64 * 2);
    float* xrec1    = (float*)walloc((size_t)N * 4 * 4);
    float* hraw     = (float*)walloc(4 * 256 * 4);
    float* gbuf     = (float*)walloc(4 * 256 * 4);
    float* data0    = (float*)walloc((size_t)N * 64 * 4);
    u16*   cat2     = (u16*)walloc((size_t)E * 128 * 2);
    u16*   xf6      = (u16*)walloc((size_t)E * 384 * 2);
    u16*   h1b      = (u16*)walloc((size_t)E * 64 * 2);
    u16*   h2b      = (u16*)walloc((size_t)E * 64 * 2);
    u16*   a1mid    = (u16*)walloc((size_t)E * 64 * 2);

    const u32 ENC_NEG_INF = 0x007FFFFFu;       // enc_ord(-inf)

    // -------- VAE reconstruction branch
    k_fill_u32<<<cdiv_h(1024, 256), 256, 0, stream>>>((u32*)hraw, 0u, 1024);
    k_vae_fc1<<<256, 256, 0, stream>>>(x, fc1_W, hraw, N);
    k_vae_head<<<1, 256, 0, stream>>>(hraw, fc1_b, fc2_W, fc2_b, fc3_W, fc3_b, fc4_W, fc4_b, gbuf);
    k_vae_rec<<<cdiv_h(N, 256), 256, 0, stream>>>(gbuf, fc5_W, fc5_b, x, out_rec, xrec1, N);
    // data0 = gelu(x_rec1 @ node_lin_W + b)
    k_lin_small<<<cdiv_h((size_t)N * 64, 256), 256, 0, stream>>>(
        xrec1, N, 4, node_lin_W, node_lin_b, 1, 0.f, nullptr, data0);
    // attr1 (bf16, shared by all conv layers)
    k_lin_small<<<cdiv_h((size_t)E * 64, 256), 256, 0, stream>>>(
        attr, E, 16, attr_lin_W, attr_lin_b, 0, 0.f, attr1_bf, nullptr);

    // -------- two GATv2 stacks (3 layers each)
    const float* pres[2] = {node_emb, data0};
    float* outs[2] = {ne_cur, data_cur};
    for (int s = 0; s < 2; ++s) {
        const float* pre = pres[s];
        const float* cur = pre;
        for (int i = 0; i < 3; ++i) {
            k_f2bf<<<cdiv_h((size_t)N * 64, 256), 256, 0, stream>>>(cur, nbf, (size_t)N * 64);
            // node transforms (WMMA, K=64)
            wmma_gemm_kernel<float, 0, 4><<<cdiv_h(N, 128), 256, 64 * 64 * 2, stream>>>(
                nbf, N, 64, cW[s][0] + (size_t)i * 4096, cBl[s] + i * 64, 64, xl, 64, 0);
            wmma_gemm_kernel<float, 0, 4><<<cdiv_h(N, 128), 256, 64 * 64 * 2, stream>>>(
                nbf, N, 64, cW[s][1] + (size_t)i * 4096, cBr[s] + i * 64, 64, xr, 64, 0);
            // edge attr transform (WMMA, dominant GEMM: E x 64 x 64)
            wmma_gemm_kernel<float, 0, 4><<<cdiv_h(E, 128), 256, 64 * 64 * 2, stream>>>(
                attr1_bf, E, 64, cW[s][2] + (size_t)i * 4096, nullptr, 64, eattr, 64, 0);
            // segment softmax + scatter
            k_fill_u32<<<cdiv_h(N, 256), 256, 0, stream>>>(nmax, ENC_NEG_INF, (size_t)N);
            k_fill_u32<<<cdiv_h(N, 256), 256, 0, stream>>>((u32*)nsum, 0u, (size_t)N);
            k_fill_u32<<<cdiv_h((size_t)N * 64, 256), 256, 0, stream>>>((u32*)nacc, 0u, (size_t)N * 64);
            k_message<<<cdiv_h(E, 256), 256, 0, stream>>>(xl, xr, eattr, src, dst,
                                                          cAtt[s] + i * 64, edge_s, nmax, E);
            k_expsum<<<cdiv_h(E, 256), 256, 0, stream>>>(edge_s, dst, nmax, nsum, E);
            k_scatter<<<cdiv_h((size_t)E * 64, 256), 256, 0, stream>>>(
                edge_s, xl, src, dst, nsum, nacc, E);
            k_combine<<<cdiv_h((size_t)N * 64, 256), 256, 0, stream>>>(
                nacc, cBias[s] + i * 64, pre, outs[s], N);
            cur = outs[s];
        }
    }

    // -------- final per-edge feature assembly: xf6 [E,384] bf16
    k_gather_misc<<<cdiv_h((size_t)E * 64, 256), 256, 0, stream>>>(
        index_t, cur_t, cur_w, embed, ttab, wtab, xf6, E);
    // a1 = leaky_relu(attr@W1+b1,0.01) @ W2 + b2  -> xf6 cols [64,128)
    k_lin_small<<<cdiv_h((size_t)E * 64, 256), 256, 0, stream>>>(
        attr, E, 16, attr1_W1, attr1_b1, 2, 0.01f, a1mid, nullptr);
    wmma_gemm_kernel<u16, 0, 4><<<cdiv_h(E, 128), 256, 64 * 64 * 2, stream>>>(
        a1mid, E, 64, attr1_W2, attr1_b2, 64, xf6, 384, 64);
    // xg = cat(data[src],data[dst]) @ gcn1 -> cols [128,192)
    k_gather_cat2<<<cdiv_h((size_t)E * 128, 256), 256, 0, stream>>>(data_cur, src, dst, cat2, E);
    wmma_gemm_kernel<u16, 0, 4><<<cdiv_h(E, 128), 256, 64 * 128 * 2, stream>>>(
        cat2, E, 128, gcn1_W, gcn1_b, 64, xf6, 384, 128);
    // xg1 = cat(ne[src],ne[dst]) @ gcn2 -> cols [192,256)
    k_gather_cat2<<<cdiv_h((size_t)E * 128, 256), 256, 0, stream>>>(ne_cur, src, dst, cat2, E);
    wmma_gemm_kernel<u16, 0, 4><<<cdiv_h(E, 128), 256, 64 * 128 * 2, stream>>>(
        cat2, E, 128, gcn2_W, gcn2_b, 64, xf6, 384, 192);

    // -------- output MLP (WMMA: K=384 -> 64 -> 64 -> 8)
    wmma_gemm_kernel<u16, 1, 4><<<cdiv_h(E, 128), 256, 64 * 384 * 2, stream>>>(
        xf6, E, 384, lins_W0, lins_b0, 64, h1b, 64, 0);
    wmma_gemm_kernel<u16, 1, 4><<<cdiv_h(E, 128), 256, 64 * 64 * 2, stream>>>(
        h1b, E, 64, lins_W1, lins_b1, 64, h2b, 64, 0);
    wmma_gemm_kernel<float, 0, 1><<<cdiv_h(E, 128), 256, 16 * 64 * 2, stream>>>(
        h2b, E, 64, lins_W2, lins_b2, 8, out_xf, 8, 0);

    (void)n_in; (void)out_size; (void)ws_size;
}
